// LiftedLoss_44169443672248
// MI455X (gfx1250) — compile-verified, matrix operand order checked by reading, not run on hardware
//
#include <hip/hip_runtime.h>
#include <hip/hip_bf16.h>
#include <math.h>

typedef __bf16 bf16_t;
typedef __attribute__((ext_vector_type(16))) __bf16 v16bf;
typedef __attribute__((ext_vector_type(8)))  __bf16 v8bf;
typedef __attribute__((ext_vector_type(8)))  float  v8f;

#define N_PTS    8192
#define DIMS     128
#define TILE     128
#define NTILES   (N_PTS / TILE)               // 64
#define NTRI     (NTILES * (NTILES + 1) / 2)  // 2080 triangular tiles
#define MARGIN_F 1.0f
#define EPS_F    1e-16f

typedef __attribute__((address_space(3))) unsigned char lds_uchar;

// CDNA5 async copy: 16 bytes global -> LDS per lane, tracked by ASYNCcnt.
__device__ __forceinline__ void async_copy_b128(void* lds_dst, const void* gsrc) {
    lds_uchar* ldst = (lds_uchar*)lds_dst;   // generic -> LDS addrspace (32-bit VGPR)
    asm volatile("global_load_async_to_lds_b128 %0, %1, off"
                 :: "v"(ldst), "v"(gsrc)
                 : "memory");
}

__device__ __forceinline__ void wait_async0() {
    asm volatile("s_wait_asynccnt 0" ::: "memory");
}

// 16x32 bf16 WMMA operand fragment (A layout; B of X*X^T uses the same
// pattern because B's source is X^T of row-major X). Works for global or LDS.
// Lane l (0..15):  row M=l,    K = kb*32 + {0..7, 16..23}
// Lane l (16..31): row M=l-16, K = kb*32 + {8..15, 24..31}
__device__ __forceinline__ v16bf load_frag(const bf16_t* __restrict__ base,
                                           int row0, int kb, int lane) {
    int m    = lane & 15;
    int koff = kb * 32 + ((lane & 16) ? 8 : 0);
    const bf16_t* p = base + (size_t)(row0 + m) * DIMS + koff;
    v8bf lo = *(const v8bf*)p;
    v8bf hi = *(const v8bf*)(p + 16);
    v16bf f;
#pragma unroll
    for (int t = 0; t < 8; ++t) { f[t] = lo[t]; f[t + 8] = hi[t]; }
    return f;
}

__device__ __forceinline__ v8f wmma_bf16(v16bf a, v16bf b, v8f c) {
    return __builtin_amdgcn_wmma_f32_16x16x32_bf16(
        false, a, false, b, (short)0, c, false, false);
}

// Fast hardware transcendentals (v_sqrt_f32 / v_log_f32 class, ~1 ulp).
__device__ __forceinline__ float fast_sqrt(float x) { return __builtin_amdgcn_sqrtf(x); }

// Decode linear triangular index t -> (bi, bj), bj <= bi.
__device__ __forceinline__ void tri_decode(int t, int& bi, int& bj) {
    bi = (int)((sqrtf(8.0f * (float)t + 1.0f) - 1.0f) * 0.5f);
    while ((bi + 1) * (bi + 2) / 2 <= t) ++bi;
    while (bi * (bi + 1) / 2 > t) --bi;
    bj = t - bi * (bi + 1) / 2;
}

// ---------------- prep: bf16 hi/lo split, squared norms, zero S/acc ----------
__global__ __launch_bounds__(128)
void lifted_prep(const float* __restrict__ feat,
                 bf16_t* __restrict__ fhi, bf16_t* __restrict__ flo,
                 float* __restrict__ sq, float* __restrict__ S,
                 float* __restrict__ acc) {
    int row = blockIdx.x;
    int t   = threadIdx.x;
    __shared__ float ssum;
    if (t == 0) ssum = 0.0f;
    __syncthreads();
    size_t idx = (size_t)row * DIMS + t;
    float x = feat[idx];
    bf16_t h = (bf16_t)x;
    float  hf = (float)h;
    bf16_t l = (bf16_t)(x - hf);
    fhi[idx] = h;
    flo[idx] = l;
    atomicAdd(&ssum, x * x);
    __syncthreads();
    if (t == 0) { sq[row] = ssum; S[row] = 0.0f; }
    if (row == 0 && t < 2) acc[t] = 0.0f;
}

// ---------------- pass 1: S_i = sum_j exp(margin - d_ij) * [lab_i != lab_j] --
// Triangular tiles. Column sums (S over J range) reduce in-lane; row sums
// (S over I range, off-diagonal tiles only) accumulate in registers across
// all column subtiles, then reduce once per row via shfl_xor at tile end.
__global__ __launch_bounds__(256)
void lifted_pass1(const bf16_t* __restrict__ fhi, const bf16_t* __restrict__ flo,
                  const float* __restrict__ sq, const int* __restrict__ labels,
                  float* __restrict__ S) {
    __shared__ __align__(16) bf16_t Bh[64 * DIMS];   // 16 KB (half of B tile, hi)
    __shared__ __align__(16) bf16_t Bl[64 * DIMS];   // 16 KB (half of B tile, lo)
    __shared__ float sqI[TILE], sqJ[TILE], SpartI[TILE], SpartJ[TILE];
    __shared__ int   labI[TILE], labJ[TILE];

    int bi, bj;
    tri_decode(blockIdx.x, bi, bj);
    const int  Ibase = bi * TILE;
    const int  Jbase = bj * TILE;
    const bool diag  = (bi == bj);

    const int tid  = threadIdx.x;
    const int wave = tid >> 5;
    const int lane = tid & 31;

    if (tid < TILE) {
        sqI[tid]  = sq[Ibase + tid];
        sqJ[tid]  = sq[Jbase + tid];
        labI[tid] = labels[Ibase + tid];
        labJ[tid] = labels[Jbase + tid];
        SpartI[tid] = 0.0f;
        SpartJ[tid] = 0.0f;
    }
    __syncthreads();

    // Hoist A fragments for this wave's 16-row strip (4 K-blocks, hi+lo).
    v16bf ah[4], al[4];
#pragma unroll
    for (int kb = 0; kb < 4; ++kb) {
        ah[kb] = load_frag(fhi, Ibase + wave * 16, kb, lane);
        al[kb] = load_frag(flo, Ibase + wave * 16, kb, lane);
    }

    const int n  = lane & 15;
    const int hh = (lane & 16) ? 1 : 0;

    float rsum[8] = {0.f, 0.f, 0.f, 0.f, 0.f, 0.f, 0.f, 0.f};

#pragma unroll 1
    for (int half = 0; half < 2; ++half) {
        // Async-stage 64 rows of B (hi+lo) into LDS: 2048 x b128, 8 per thread.
        const bf16_t* srcH = fhi + (size_t)(Jbase + half * 64) * DIMS;
        const bf16_t* srcL = flo + (size_t)(Jbase + half * 64) * DIMS;
#pragma unroll
        for (int c = 0; c < 4; ++c) {
            int seg = c * 256 + tid;          // 0..1023
            int jr  = seg >> 4;
            int o   = (seg & 15) * 8;
            async_copy_b128(&Bh[jr * DIMS + o], srcH + (size_t)jr * DIMS + o);
            async_copy_b128(&Bl[jr * DIMS + o], srcL + (size_t)jr * DIMS + o);
        }
        wait_async0();
        __syncthreads();

#pragma unroll 1
        for (int jsh = 0; jsh < 4; ++jsh) {
            const int js = half * 4 + jsh;
            v8f acc = {0.f, 0.f, 0.f, 0.f, 0.f, 0.f, 0.f, 0.f};
#pragma unroll
            for (int kb = 0; kb < 4; ++kb) {
                v16bf bh = load_frag(Bh, jsh * 16, kb, lane);
                v16bf bl = load_frag(Bl, jsh * 16, kb, lane);
                acc = wmma_bf16(ah[kb], bh, acc);   // hi*hi
                acc = wmma_bf16(ah[kb], bl, acc);   // hi*lo
                acc = wmma_bf16(al[kb], bh, acc);   // lo*hi
            }
            const int   j_l = js * 16 + n;
            const float sqj = sqJ[j_l];
            const int   lj  = labJ[j_l];
            float colsum = 0.0f;
#pragma unroll
            for (int r = 0; r < 8; ++r) {
                int   i_l  = wave * 16 + r + 8 * hh;
                float g    = acc[r];
                float d2   = sqI[i_l] + sqj - 2.0f * g;
                float dist = fast_sqrt(fmaxf(d2, EPS_F));
                float wv   = (labI[i_l] != lj) ? __expf(MARGIN_F - dist) : 0.0f;
                colsum  += wv;
                rsum[r] += wv;
            }
            atomicAdd(&SpartJ[j_l], colsum);
        }
        __syncthreads();   // LDS reuse / Spart visibility
    }

    // Deferred row-sum reduction (off-diagonal tiles only): one 4-step
    // shfl_xor tree per row instead of per element.
    if (!diag) {
#pragma unroll
        for (int r = 0; r < 8; ++r) {
            float rs = rsum[r];
            rs += __shfl_xor(rs, 1, 16);
            rs += __shfl_xor(rs, 2, 16);
            rs += __shfl_xor(rs, 4, 16);
            rs += __shfl_xor(rs, 8, 16);
            if (n == 0) atomicAdd(&SpartI[wave * 16 + r + 8 * hh], rs);
        }
    }
    __syncthreads();

    if (tid < TILE) {
        atomicAdd(&S[Jbase + tid], SpartJ[tid]);
        if (!diag) atomicAdd(&S[Ibase + tid], SpartI[tid]);
    }
}

// ---------------- pass 2: loss over positive pairs (triangular, weight 2) ----
__global__ __launch_bounds__(256)
void lifted_pass2(const bf16_t* __restrict__ fhi, const bf16_t* __restrict__ flo,
                  const float* __restrict__ sq, const int* __restrict__ labels,
                  const float* __restrict__ S, float* __restrict__ gacc) {
    __shared__ __align__(16) bf16_t Bh[64 * DIMS];
    __shared__ __align__(16) bf16_t Bl[64 * DIMS];
    __shared__ float sqI[TILE], sqJ[TILE], SI[TILE], SJ[TILE];
    __shared__ int   labI[TILE], labJ[TILE];
    __shared__ float shLoss, shCnt;

    int bi, bj;
    tri_decode(blockIdx.x, bi, bj);
    const int   Ibase = bi * TILE;
    const int   Jbase = bj * TILE;
    const float wgt   = (bi == bj) ? 1.0f : 2.0f;

    const int tid  = threadIdx.x;
    const int wave = tid >> 5;
    const int lane = tid & 31;

    if (tid < TILE) {
        sqI[tid]  = sq[Ibase + tid];
        sqJ[tid]  = sq[Jbase + tid];
        SI[tid]   = S[Ibase + tid];
        SJ[tid]   = S[Jbase + tid];
        labI[tid] = labels[Ibase + tid];
        labJ[tid] = labels[Jbase + tid];
    }
    if (tid == 0) { shLoss = 0.0f; shCnt = 0.0f; }
    __syncthreads();

    v16bf ah[4], al[4];
#pragma unroll
    for (int kb = 0; kb < 4; ++kb) {
        ah[kb] = load_frag(fhi, Ibase + wave * 16, kb, lane);
        al[kb] = load_frag(flo, Ibase + wave * 16, kb, lane);
    }

    const int n  = lane & 15;
    const int hh = (lane & 16) ? 1 : 0;
    float lsum = 0.0f, csum = 0.0f;

#pragma unroll 1
    for (int half = 0; half < 2; ++half) {
        const bf16_t* srcH = fhi + (size_t)(Jbase + half * 64) * DIMS;
        const bf16_t* srcL = flo + (size_t)(Jbase + half * 64) * DIMS;
#pragma unroll
        for (int c = 0; c < 4; ++c) {
            int seg = c * 256 + tid;
            int jr  = seg >> 4;
            int o   = (seg & 15) * 8;
            async_copy_b128(&Bh[jr * DIMS + o], srcH + (size_t)jr * DIMS + o);
            async_copy_b128(&Bl[jr * DIMS + o], srcL + (size_t)jr * DIMS + o);
        }
        wait_async0();
        __syncthreads();

#pragma unroll 1
        for (int jsh = 0; jsh < 4; ++jsh) {
            const int js = half * 4 + jsh;
            v8f acc = {0.f, 0.f, 0.f, 0.f, 0.f, 0.f, 0.f, 0.f};
#pragma unroll
            for (int kb = 0; kb < 4; ++kb) {
                v16bf bh = load_frag(Bh, jsh * 16, kb, lane);
                v16bf bl = load_frag(Bl, jsh * 16, kb, lane);
                acc = wmma_bf16(ah[kb], bh, acc);
                acc = wmma_bf16(ah[kb], bl, acc);
                acc = wmma_bf16(al[kb], bh, acc);
            }
            const int   j_l = js * 16 + n;
            const float sqj = sqJ[j_l];
            const float Sj  = SJ[j_l];
            const int   lj  = labJ[j_l];
#pragma unroll
            for (int r = 0; r < 8; ++r) {
                int   i_l  = wave * 16 + r + 8 * hh;
                float g    = acc[r];
                float d2   = sqI[i_l] + sqj - 2.0f * g;
                float dist = fast_sqrt(fmaxf(d2, EPS_F));
                if (labI[i_l] == lj) {
                    float Jv = __logf(SI[i_l] + Sj) + dist;
                    float h  = fmaxf(Jv, 0.0f);
                    lsum += h * h;
                    csum += 1.0f;
                }
            }
        }
        __syncthreads();
    }

    atomicAdd(&shLoss, lsum * wgt);
    atomicAdd(&shCnt,  csum * wgt);
    __syncthreads();
    if (tid == 0) {
        atomicAdd(&gacc[0], shLoss);
        atomicAdd(&gacc[1], shCnt);
    }
}

// ---------------- finalize ---------------------------------------------------
__global__ void lifted_finalize(const float* __restrict__ gacc,
                                float* __restrict__ out) {
    out[0] = gacc[0] / (2.0f * gacc[1]);
}

extern "C" void kernel_launch(void* const* d_in, const int* in_sizes, int n_in,
                              void* d_out, int out_size, void* d_ws, size_t ws_size,
                              hipStream_t stream) {
    const float* feat   = (const float*)d_in[0];
    const int*   labels = (const int*)d_in[1];

    bf16_t* fhi = (bf16_t*)d_ws;                          // N*D bf16 (2 MB)
    bf16_t* flo = fhi + (size_t)N_PTS * DIMS;             // N*D bf16 (2 MB)
    float*  sq  = (float*)(flo + (size_t)N_PTS * DIMS);   // N f32
    float*  S   = sq + N_PTS;                             // N f32
    float*  acc = S + N_PTS;                              // 2 f32

    lifted_prep<<<N_PTS, 128, 0, stream>>>(feat, fhi, flo, sq, S, acc);
    lifted_pass1<<<NTRI, 256, 0, stream>>>(fhi, flo, sq, labels, S);
    lifted_pass2<<<NTRI, 256, 0, stream>>>(fhi, flo, sq, labels, S, acc);
    lifted_finalize<<<1, 1, 0, stream>>>(acc, (float*)d_out);
}
